// GNN_7782480740940
// MI455X (gfx1250) — matrix operand order repaired and checked by reading, not compiled, not run on hardware
//
#include <hip/hip_runtime.h>
#include <hip/hip_bf16.h>
#include <math.h>

// ---------------------------------------------------------------------------
// GNN pipeline for MI455X (gfx1250, wave32, WMMA bf16)
//   N=20000 nodes, E=200000 edges, HID=32, IN=64, OUT=64, 3 GRU steps
// ---------------------------------------------------------------------------

#define Nn_    20000
#define E_     200000
#define HID_   32
#define IN_    64
#define OUT_   64
#define MLPH_  128      // edge MLP hidden
#define WSQ_   1024     // HID*HID

typedef __attribute__((ext_vector_type(16))) __bf16        v16bf;
typedef __attribute__((ext_vector_type(8)))  float         v8f;
typedef __attribute__((ext_vector_type(4)))  unsigned int  u32x4;

union U16 { v16bf v; u32x4 q[2]; };

// fast tanh: tanh(x) = 1 - 2/(exp(2x)+1); exact saturation at +/-inf, NaN-safe.
// 1x v_exp_f32 + 1x v_rcp_f32 + 2 FMA-class ops (vs ~20-op libm tanhf).
__device__ __forceinline__ float fast_tanh(float x) {
    float e = __expf(2.0f * x);
    return 1.0f - 2.0f * __builtin_amdgcn_rcpf(e + 1.0f);
}
__device__ __forceinline__ float fast_sigmoid(float x) {
    return __builtin_amdgcn_rcpf(1.0f + __expf(-x));
}

// ---------------------------------------------------------------------------
// Pack W1 (32x128 f32) into WMMA B-operand bf16 layout.
// B tile (k0,nt): lane l holds N = nt*16+(l&15), K = k0 + (l<16?0:8) + (kk<8?kk:kk+8), kk=0..15
// ---------------------------------------------------------------------------
__global__ void pack_w1(const float* __restrict__ W1, __bf16* __restrict__ W1B) {
    int tid = blockIdx.x * blockDim.x + threadIdx.x;     // 8*32*16 = 4096
    if (tid >= 8 * 32 * 16) return;
    int kk   = tid & 15;
    int lane = (tid >> 4) & 31;
    int nt   = tid >> 9;
    int K = ((lane < 16) ? 0 : 8) + (kk < 8 ? kk : kk + 8); // K in 0..31
    int Ncol = nt * 16 + (lane & 15);                    // N in 0..127
    W1B[tid] = (__bf16)W1[K * MLPH_ + Ncol];
}

// W2 (128x1024 f32) -> bf16 B-operand layout: [kt(4)][nt(64)][lane(32)][kk(16)]
__global__ void pack_w2(const float* __restrict__ W2, __bf16* __restrict__ W2B) {
    int tid = blockIdx.x * blockDim.x + threadIdx.x;     // 4*64*32*16 = 131072
    if (tid >= 4 * 64 * 32 * 16) return;
    int kk   = tid & 15;
    int lane = (tid >> 4) & 31;
    int nt   = (tid >> 9) & 63;
    int kt   = tid >> 15;
    int K = kt * 32 + ((lane < 16) ? 0 : 8) + (kk < 8 ? kk : kk + 8); // 0..127
    int Ncol = nt * 16 + (lane & 15);                    // 0..1023
    W2B[tid] = (__bf16)W2[K * WSQ_ + Ncol];
}

// ---------------------------------------------------------------------------
// h = x @ W_emb + b_emb       [N,32]   (thread per (n,k))
// ---------------------------------------------------------------------------
__global__ void embed_kernel(const float* __restrict__ x,
                             const float* __restrict__ W, const float* __restrict__ b,
                             float* __restrict__ h) {
    int idx = blockIdx.x * blockDim.x + threadIdx.x;
    int n = idx >> 5, k = idx & 31;
    if (n >= Nn_) return;
    float acc = b[k];
    const float* xr = x + (size_t)n * IN_;
#pragma unroll
    for (int d = 0; d < IN_; ++d) acc = fmaf(xr[d], W[d * HID_ + k], acc);
    h[(size_t)n * HID_ + k] = acc;
}

// ---------------------------------------------------------------------------
// he = relu((h[src]-h[dst]) @ W_ec + b_ec), stored bf16 [E,32]. Wave per edge.
// ---------------------------------------------------------------------------
__global__ void edgefeat_kernel(const float* __restrict__ h,
                                const int* __restrict__ src, const int* __restrict__ dst,
                                const float* __restrict__ Wec, const float* __restrict__ bec,
                                __bf16* __restrict__ he) {
    int gw   = (blockIdx.x * blockDim.x + threadIdx.x) >> 5;
    int lane = threadIdx.x & 31;
    if (gw >= E_) return;
    int s = src[gw], d = dst[gw];
    float dv = h[(size_t)s * HID_ + lane] - h[(size_t)d * HID_ + lane];
    float acc = bec[lane];
#pragma unroll
    for (int j = 0; j < 32; ++j) {
        float dj = __shfl(dv, j, 32);
        acc = fmaf(dj, Wec[j * HID_ + lane], acc);
    }
    acc = acc > 0.f ? acc : 0.f;
    he[(size_t)gw * HID_ + lane] = (__bf16)acc;
}

// ---------------------------------------------------------------------------
// We = tanh(relu(he@W1+b1)@W2+b2) -> bf16 [E,1024].  WMMA bf16, M=32 per wave.
// ---------------------------------------------------------------------------
#define K3_WAVES 4
__global__ void
__launch_bounds__(K3_WAVES * 32)
edge_weight_gemm(const __bf16* __restrict__ he,
                 const __bf16* __restrict__ W1B, const float* __restrict__ b1,
                 const __bf16* __restrict__ W2B, const float* __restrict__ b2,
                 __bf16* __restrict__ We) {
    __shared__ __align__(16) __bf16 Tsm[K3_WAVES * 2 * 16 * MLPH_];   // 32 KB

    const int wid  = threadIdx.x >> 5;
    const int lane = threadIdx.x & 31;
    const int t32  = blockIdx.x * K3_WAVES + wid;           // tile of 32 edges
    const bool active = (t32 < (E_ / 32));

    __bf16* T = &Tsm[wid * (2 * 16 * MLPH_)];
    const int row_lo = lane & 15;
    const int kb     = (lane < 16) ? 0 : 8;
    const int mrow   = (lane < 16) ? 0 : 8;
    const int col_lo = lane & 15;

    // -------- stage 1: T = bf16(relu(he_tile @ W1 + b1))  [2 x 16 x 128]
    if (active) {
        const int e0 = t32 * 32;
#pragma unroll
        for (int m = 0; m < 2; ++m) {
            U16 A;
            const __bf16* hrow = he + (size_t)(e0 + m * 16 + row_lo) * HID_;
            A.q[0] = *(const u32x4*)(hrow + kb);
            A.q[1] = *(const u32x4*)(hrow + kb + 16);
#pragma unroll
            for (int nt = 0; nt < 8; ++nt) {
                U16 B;
                const __bf16* bp = W1B + ((size_t)nt * 32 + lane) * 16;
                B.q[0] = *(const u32x4*)(bp);
                B.q[1] = *(const u32x4*)(bp + 8);
                v8f acc = {};
                acc = __builtin_amdgcn_wmma_f32_16x16x32_bf16(
                        false, A.v, false, B.v, (short)0, acc, false, false);
                int col = nt * 16 + col_lo;
                float bias = b1[col];
#pragma unroll
                for (int r = 0; r < 8; ++r) {
                    float v = acc[r] + bias;
                    v = v > 0.f ? v : 0.f;
                    T[m * 2048 + (mrow + r) * MLPH_ + col] = (__bf16)v;
                }
            }
        }
    }
    __syncthreads();

    // -------- stage 2: We_tile = tanh(T @ W2 + b2)  [32 x 1024]
    if (active) {
        const int e0 = t32 * 32;
        U16 A2[2][4];
#pragma unroll
        for (int m = 0; m < 2; ++m)
#pragma unroll
            for (int kt = 0; kt < 4; ++kt) {
                const __bf16* tp = &T[m * 2048 + row_lo * MLPH_ + kt * 32 + kb];
                A2[m][kt].q[0] = *(const u32x4*)(tp);
                A2[m][kt].q[1] = *(const u32x4*)(tp + 16);
            }

        for (int nt = 0; nt < 64; ++nt) {
            v8f acc0 = {}, acc1 = {};
#pragma unroll
            for (int kt = 0; kt < 4; ++kt) {
                U16 B;
                const __bf16* bp = W2B + (((size_t)kt * 64 + nt) * 32 + lane) * 16;
                B.q[0] = *(const u32x4*)(bp);
                B.q[1] = *(const u32x4*)(bp + 8);
                acc0 = __builtin_amdgcn_wmma_f32_16x16x32_bf16(
                         false, A2[0][kt].v, false, B.v, (short)0, acc0, false, false);
                acc1 = __builtin_amdgcn_wmma_f32_16x16x32_bf16(
                         false, A2[1][kt].v, false, B.v, (short)0, acc1, false, false);
            }
            int ncol = nt * 16 + col_lo;
            float bias = b2[ncol];
#pragma unroll
            for (int r = 0; r < 8; ++r) {
                float v0 = fast_tanh(acc0[r] + bias);
                float v1 = fast_tanh(acc1[r] + bias);
                We[(size_t)(e0 + mrow + r) * WSQ_ + ncol]      = (__bf16)v0;
                We[(size_t)(e0 + 16 + mrow + r) * WSQ_ + ncol] = (__bf16)v1;
            }
        }
    }
}

// ---------------------------------------------------------------------------
// zero accumulator
// ---------------------------------------------------------------------------
__global__ void zero_kernel(float* __restrict__ p, int n) {
    int i = blockIdx.x * blockDim.x + threadIdx.x;
    if (i < n) p[i] = 0.f;
}

// ---------------------------------------------------------------------------
// m = We[e] @ h[src[e]]; atomic scatter-add into a[dst].  Wave per edge.
// lane i holds row i of We[e] (32 bf16 = 64B, two clause-batched b128 pairs).
// ---------------------------------------------------------------------------
__global__ void matvec_scatter(const __bf16* __restrict__ We,
                               const float* __restrict__ h,
                               const int* __restrict__ src, const int* __restrict__ dst,
                               float* __restrict__ a) {
    int gw   = (blockIdx.x * blockDim.x + threadIdx.x) >> 5;
    int lane = threadIdx.x & 31;
    if (gw >= E_) return;
    int s = src[gw], d = dst[gw];
    float hv = h[(size_t)s * HID_ + lane];

    const __bf16* wrow = We + (size_t)gw * WSQ_ + lane * 32;
    U16 w0, w1;
    w0.q[0] = *(const u32x4*)(wrow);
    w0.q[1] = *(const u32x4*)(wrow + 8);
    w1.q[0] = *(const u32x4*)(wrow + 16);
    w1.q[1] = *(const u32x4*)(wrow + 24);

    float m = 0.f;
#pragma unroll
    for (int j = 0; j < 32; ++j) {
        float hj = __shfl(hv, j, 32);
        float wv = (j < 16) ? (float)w0.v[j] : (float)w1.v[j - 16];
        m = fmaf(wv, hj, m);
    }
    atomicAdd(&a[(size_t)d * HID_ + lane], m);
}

// ---------------------------------------------------------------------------
// GRU node update. Wave per node, lane = feature k.
// ---------------------------------------------------------------------------
__global__ void gru_kernel(const float* __restrict__ a, const float* __restrict__ h,
                           const float* __restrict__ Wi, const float* __restrict__ bi,
                           const float* __restrict__ Wh, const float* __restrict__ bh,
                           float* __restrict__ hout) {
    int gw   = (blockIdx.x * blockDim.x + threadIdx.x) >> 5;
    int lane = threadIdx.x & 31;
    if (gw >= Nn_) return;
    float av = a[(size_t)gw * HID_ + lane];
    float hv = h[(size_t)gw * HID_ + lane];

    float ir = bi[lane], iz = bi[32 + lane], in_ = bi[64 + lane];
    float hr = bh[lane], hz = bh[32 + lane], hn  = bh[64 + lane];
#pragma unroll
    for (int j = 0; j < 32; ++j) {
        float aj = __shfl(av, j, 32);
        float hj = __shfl(hv, j, 32);
        const float* wi = Wi + j * 96;
        const float* wh = Wh + j * 96;
        ir  = fmaf(aj, wi[lane],      ir);
        iz  = fmaf(aj, wi[32 + lane], iz);
        in_ = fmaf(aj, wi[64 + lane], in_);
        hr  = fmaf(hj, wh[lane],      hr);
        hz  = fmaf(hj, wh[32 + lane], hz);
        hn  = fmaf(hj, wh[64 + lane], hn);
    }
    float r = fast_sigmoid(ir + hr);
    float z = fast_sigmoid(iz + hz);
    float n = fast_tanh(in_ + r * hn);
    hout[(size_t)gw * HID_ + lane] = (1.f - z) * n + z * hv;
}

// ---------------------------------------------------------------------------
// out = tanh(h @ W_out + b_out)   [N,64]  (thread per (n,o))
// ---------------------------------------------------------------------------
__global__ void out_kernel(const float* __restrict__ h,
                           const float* __restrict__ W, const float* __restrict__ b,
                           float* __restrict__ out) {
    int idx = blockIdx.x * blockDim.x + threadIdx.x;
    int n = idx >> 6, o = idx & 63;
    if (n >= Nn_) return;
    float acc = b[o];
    const float* hr = h + (size_t)n * HID_;
#pragma unroll
    for (int j = 0; j < HID_; ++j) acc = fmaf(hr[j], W[j * OUT_ + o], acc);
    out[(size_t)n * OUT_ + o] = fast_tanh(acc);
}

// ---------------------------------------------------------------------------
extern "C" void kernel_launch(void* const* d_in, const int* in_sizes, int n_in,
                              void* d_out, int out_size, void* d_ws, size_t ws_size,
                              hipStream_t stream) {
    const float* x     = (const float*)d_in[0];
    const int*   src   = (const int*)  d_in[1];
    const int*   dst   = (const int*)  d_in[2];
    const float* W_emb = (const float*)d_in[3];
    const float* b_emb = (const float*)d_in[4];
    const float* W_ec  = (const float*)d_in[5];
    const float* b_ec  = (const float*)d_in[6];
    const float* W1    = (const float*)d_in[7];
    const float* b1    = (const float*)d_in[8];
    const float* W2    = (const float*)d_in[9];
    const float* b2    = (const float*)d_in[10];
    const float* W_i   = (const float*)d_in[11];
    const float* b_i   = (const float*)d_in[12];
    const float* W_h   = (const float*)d_in[13];
    const float* b_h   = (const float*)d_in[14];
    const float* W_out = (const float*)d_in[15];
    const float* b_out = (const float*)d_in[16];
    float* out = (float*)d_out;

    // ---- workspace carve (256B aligned) ----
    char* ws = (char*)d_ws;
    size_t off = 0;
    auto carve = [&](size_t bytes) { size_t o = off; off = (off + bytes + 255) & ~(size_t)255; return o; };
    float*  h0  = (float*) (ws + carve((size_t)Nn_ * HID_ * 4));
    float*  h1  = (float*) (ws + carve((size_t)Nn_ * HID_ * 4));
    float*  acc = (float*) (ws + carve((size_t)Nn_ * HID_ * 4));
    __bf16* he  = (__bf16*)(ws + carve((size_t)E_ * HID_ * 2));
    __bf16* W1B = (__bf16*)(ws + carve((size_t)8 * 32 * 16 * 2));
    __bf16* W2B = (__bf16*)(ws + carve((size_t)4 * 64 * 32 * 16 * 2));
    __bf16* We  = (__bf16*)(ws + carve((size_t)E_ * WSQ_ * 2));
    (void)ws_size; (void)n_in; (void)in_sizes; (void)out_size;

    // ---- pack weights into WMMA B-operand layout ----
    pack_w1<<<(4096 + 255) / 256, 256, 0, stream>>>(W1, W1B);
    pack_w2<<<(131072 + 255) / 256, 256, 0, stream>>>(W2, W2B);

    // ---- node embedding ----
    embed_kernel<<<(Nn_ * 32 + 255) / 256, 256, 0, stream>>>(x, W_emb, b_emb, h0);

    // ---- edge features ----
    edgefeat_kernel<<<(E_ * 32 + 255) / 256, 256, 0, stream>>>(h0, src, dst, W_ec, b_ec, he);

    // ---- per-edge weight matrices via WMMA (the 52 GFLOP GEMM) ----
    {
        int tiles32 = E_ / 32;                                   // 6250
        int blocks  = (tiles32 + K3_WAVES - 1) / K3_WAVES;       // 1563
        edge_weight_gemm<<<blocks, K3_WAVES * 32, 0, stream>>>(he, W1B, b1, W2B, b2, We);
    }

    // ---- 3 gated propagation steps (ping-pong h) ----
    float* hc = h0;
    float* hn = h1;
    for (int step = 0; step < 3; ++step) {
        zero_kernel<<<(Nn_ * HID_ + 255) / 256, 256, 0, stream>>>(acc, Nn_ * HID_);
        matvec_scatter<<<(E_ * 32 + 255) / 256, 256, 0, stream>>>(We, hc, src, dst, acc);
        gru_kernel<<<(Nn_ * 32 + 255) / 256, 256, 0, stream>>>(acc, hc, W_i, b_i, W_h, b_h, hn);
        float* t = hc; hc = hn; hn = t;
    }

    // ---- output head (hc holds final h) ----
    out_kernel<<<(Nn_ * 64 + 255) / 256, 256, 0, stream>>>(hc, W_out, b_out, out);
}
